// GCNNet_89833535963136
// MI455X (gfx1250) — compile-verified, matrix operand order checked by reading, not run on hardware
//
#include <hip/hip_runtime.h>
#include <hip/hip_bf16.h>

#define N_NODES 100000

typedef __attribute__((ext_vector_type(16))) __bf16 v16bf;
typedef __attribute__((ext_vector_type(8)))  float  v8f;

// ---------------------------------------------------------------- utility
__global__ void fill_f32(float* __restrict__ p, float v, int n) {
    int i = blockIdx.x * blockDim.x + threadIdx.x;
    int stride = gridDim.x * blockDim.x;
    for (; i < n; i += stride) p[i] = v;
}

// W1: [256][128] f32 -> W1T: [128][256] bf16 (column-of-W as contiguous K run)
__global__ void conv_w1(const float* __restrict__ W1, __bf16* __restrict__ W1T) {
    int idx = blockIdx.x * blockDim.x + threadIdx.x;   // 128*256
    if (idx < 128 * 256) {
        int n = idx >> 8, k = idx & 255;
        W1T[idx] = (__bf16)W1[k * 128 + n];
    }
}

// W2: [128][40] f32 -> W2T: [48][128] bf16 zero-padded to 48 cols
__global__ void conv_w2(const float* __restrict__ W2, __bf16* __restrict__ W2T) {
    int idx = blockIdx.x * blockDim.x + threadIdx.x;   // 48*128
    if (idx < 48 * 128) {
        int n = idx >> 7, k = idx & 127;
        W2T[idx] = (n < 40) ? (__bf16)W2[k * 40 + n] : (__bf16)0.0f;
    }
}

// deg starts at 1.0 (self loop); add 1 per incoming edge
__global__ void degree_kernel(const long long* __restrict__ ei, float* __restrict__ deg, int E) {
    int e = blockIdx.x * blockDim.x + threadIdx.x;
    if (e < E) {
        long long d = ei[(long long)E + e];
        unsafeAtomicAdd(&deg[d], 1.0f);
    }
}

__global__ void dinv_kernel(float* __restrict__ deg, int n) {
    int i = blockIdx.x * blockDim.x + threadIdx.x;
    if (i < n) deg[i] = rsqrtf(fmaxf(deg[i], 1.0f));
}

// ---------------------------------------------------------------- GEMM 1
// g1[N,128] = dinv ⊙ (x[N,256] @ W1).  Block = 8 waves = 32 rows x 128 cols.
// Wave w: row-chunk (w>>2)*16, col tiles (w&3)*32 .. +31 (two 16x16 tiles).
__global__ __launch_bounds__(256, 2)
void gemm1_wmma(const float* __restrict__ x, const __bf16* __restrict__ W1T,
                const float* __restrict__ dinv, float* __restrict__ g1) {
    const int wave = threadIdx.x >> 5;
    const int lane = threadIdx.x & 31;
    const int half = lane >> 4;          // K-half select per CDNA5 A/B layout
    const int lr   = lane & 15;
    const int off  = half ? 8 : 0;
    const int r0   = blockIdx.x * 32 + (wave >> 2) * 16;
    const int c0   = (wave & 3) * 32;
    const int row  = r0 + lr;

    v8f acc0 = {}; v8f acc1 = {};
    const float*  xr  = x   + (size_t)row * 256;
    const __bf16* wb0 = W1T + (size_t)(c0 + lr) * 256;
    const __bf16* wb1 = W1T + (size_t)(c0 + 16 + lr) * 256;

    for (int k0 = 0; k0 < 256; k0 += 32) {
        v16bf a, b0, b1;
        const float*  xp = xr  + k0 + off;
        const __bf16* p0 = wb0 + k0 + off;
        const __bf16* p1 = wb1 + k0 + off;
#pragma unroll
        for (int j = 0; j < 8; ++j) {
            a[j]      = (__bf16)xp[j];
            a[8 + j]  = (__bf16)xp[16 + j];
            b0[j]     = p0[j];      b0[8 + j] = p0[16 + j];
            b1[j]     = p1[j];      b1[8 + j] = p1[16 + j];
        }
        acc0 = __builtin_amdgcn_wmma_f32_16x16x32_bf16(false, a, false, b0, (short)0, acc0, false, false);
        acc1 = __builtin_amdgcn_wmma_f32_16x16x32_bf16(false, a, false, b1, (short)0, acc1, false, false);
    }
#pragma unroll
    for (int r = 0; r < 8; ++r) {
        int orow = r0 + r + half * 8;
        float s = dinv[orow];
        g1[(size_t)orow * 128 + c0 + lr]      = s * acc0[r];
        g1[(size_t)orow * 128 + c0 + 16 + lr] = s * acc1[r];
    }
}

// ---------------------------------------------------------------- GEMM 2
// g2[N,40] = dinv ⊙ (h1[N,128] @ W2).  Block = 3 waves, 16 rows; wave = col tile.
__global__ __launch_bounds__(96, 4)
void gemm2_wmma(const float* __restrict__ h1, const __bf16* __restrict__ W2T,
                const float* __restrict__ dinv, float* __restrict__ g2) {
    const int wave = threadIdx.x >> 5;   // 0..2
    const int lane = threadIdx.x & 31;
    const int half = lane >> 4;
    const int lr   = lane & 15;
    const int off  = half ? 8 : 0;
    const int r0   = blockIdx.x * 16;
    const int c0   = wave * 16;
    const int row  = r0 + lr;

    v8f acc = {};
    const float*  hr = h1  + (size_t)row * 128;
    const __bf16* wb = W2T + (size_t)(c0 + lr) * 128;

    for (int k0 = 0; k0 < 128; k0 += 32) {
        v16bf a, b;
        const float*  hp = hr + k0 + off;
        const __bf16* p  = wb + k0 + off;
#pragma unroll
        for (int j = 0; j < 8; ++j) {
            a[j]     = (__bf16)hp[j];
            a[8 + j] = (__bf16)hp[16 + j];
            b[j]     = p[j];
            b[8 + j] = p[16 + j];
        }
        acc = __builtin_amdgcn_wmma_f32_16x16x32_bf16(false, a, false, b, (short)0, acc, false, false);
    }
    int col = c0 + lr;
    if (col < 40) {
#pragma unroll
        for (int r = 0; r < 8; ++r) {
            int orow = r0 + r + half * 8;
            g2[(size_t)orow * 40 + col] = dinv[orow] * acc[r];
        }
    }
}

// ---------------------------------------------------------------- scatters
// One wave32 per edge: float4 per lane covers 128 features in one pass.
__global__ void scatter1(const long long* __restrict__ ei, const float* __restrict__ g1,
                         float* __restrict__ agg1, int E) {
    int gw   = (blockIdx.x * blockDim.x + threadIdx.x) >> 5;
    int lane = threadIdx.x & 31;
    int nw   = (gridDim.x * blockDim.x) >> 5;
    for (int e = gw; e < E; e += nw) {
        long long s = ei[e];
        long long d = ei[(long long)E + e];
        const float4 v = *reinterpret_cast<const float4*>(g1 + s * 128 + lane * 4);
        float* dp = agg1 + d * 128 + lane * 4;
        unsafeAtomicAdd(dp + 0, v.x);
        unsafeAtomicAdd(dp + 1, v.y);
        unsafeAtomicAdd(dp + 2, v.z);
        unsafeAtomicAdd(dp + 3, v.w);
    }
}

__global__ void scatter2(const long long* __restrict__ ei, const float* __restrict__ g2,
                         float* __restrict__ agg2, int E) {
    int gw   = (blockIdx.x * blockDim.x + threadIdx.x) >> 5;
    int lane = threadIdx.x & 31;
    int nw   = (gridDim.x * blockDim.x) >> 5;
    for (int e = gw; e < E; e += nw) {
        long long s = ei[e];
        long long d = ei[(long long)E + e];
        float v = g2[s * 40 + lane];
        unsafeAtomicAdd(agg2 + d * 40 + lane, v);
        if (lane < 8) {
            float v2 = g2[s * 40 + 32 + lane];
            unsafeAtomicAdd(agg2 + d * 40 + 32 + lane, v2);
        }
    }
}

// ---------------------------------------------------------------- finalize
// h1 = relu(dinv[d]*(g1 + agg1) + b1)  (in-place over agg1)
__global__ void finalize1(const float* __restrict__ g1, float* __restrict__ h1,
                          const float* __restrict__ b1, const float* __restrict__ dinv) {
    int idx = blockIdx.x * blockDim.x + threadIdx.x;
    if (idx < N_NODES * 128) {
        int d = idx >> 7, c = idx & 127;
        float v = dinv[d] * (g1[idx] + h1[idx]) + b1[c];
        h1[idx] = fmaxf(v, 0.0f);
    }
}

// out = log_softmax(dinv[d]*(g2 + agg2) + b2) ; one wave per node, shfl reduce
__global__ void finalize2(const float* __restrict__ g2, const float* __restrict__ agg2,
                          const float* __restrict__ b2, const float* __restrict__ dinv,
                          float* __restrict__ out) {
    int wave = threadIdx.x >> 5, lane = threadIdx.x & 31;
    int d = blockIdx.x * 8 + wave;
    if (d >= N_NODES) return;
    float s = dinv[d];
    int base = d * 40;
    float v0 = s * (g2[base + lane] + agg2[base + lane]) + b2[lane];
    float v1 = -__builtin_inff();
    if (lane < 8)
        v1 = s * (g2[base + 32 + lane] + agg2[base + 32 + lane]) + b2[32 + lane];
    float m = fmaxf(v0, v1);
#pragma unroll
    for (int o = 16; o >= 1; o >>= 1) m = fmaxf(m, __shfl_xor(m, o, 32));
    float sum = __expf(v0 - m) + ((lane < 8) ? __expf(v1 - m) : 0.0f);
#pragma unroll
    for (int o = 16; o >= 1; o >>= 1) sum += __shfl_xor(sum, o, 32);
    float lse = __logf(sum);
    out[base + lane] = v0 - m - lse;
    if (lane < 8) out[base + 32 + lane] = v1 - m - lse;
}

// ---------------------------------------------------------------- launcher
extern "C" void kernel_launch(void* const* d_in, const int* in_sizes, int n_in,
                              void* d_out, int out_size, void* d_ws, size_t ws_size,
                              hipStream_t stream) {
    const float*     x  = (const float*)d_in[0];
    const long long* ei = (const long long*)d_in[1];   // int64 [2,E]
    const float*     W1 = (const float*)d_in[2];
    const float*     b1 = (const float*)d_in[3];
    const float*     W2 = (const float*)d_in[4];
    const float*     b2 = (const float*)d_in[5];
    float* out = (float*)d_out;
    const int E = in_sizes[1] / 2;

    char* ws = (char*)d_ws;
    // workspace map (bytes):
    float*  dinv = (float*)(ws);                                  // 400 KB
    __bf16* W1T  = (__bf16*)(ws + (size_t)512 * 1024);            // 64 KB
    __bf16* W2T  = (__bf16*)(ws + (size_t)640 * 1024);            // 12 KB
    float*  g1   = (float*)(ws + (size_t)1  * 1024 * 1024);       // 51.2 MB
    float*  h1   = (float*)(ws + (size_t)54 * 1024 * 1024);       // agg1 -> h1, 51.2 MB
    float*  g2   = (float*)(ws + (size_t)1  * 1024 * 1024);       // reuse g1 region, 16 MB
    float*  agg2 = (float*)(ws + (size_t)21 * 1024 * 1024);       // 16 MB (inside g1 region)

    // deg init = 1.0 (self loop), agg1 = 0
    fill_f32<<<1024, 256, 0, stream>>>(dinv, 1.0f, N_NODES);
    fill_f32<<<2048, 256, 0, stream>>>(h1, 0.0f, N_NODES * 128);
    conv_w1<<<(128 * 256 + 255) / 256, 256, 0, stream>>>(W1, W1T);
    conv_w2<<<(48 * 128 + 255) / 256, 256, 0, stream>>>(W2, W2T);

    degree_kernel<<<(E + 255) / 256, 256, 0, stream>>>(ei, dinv, E);
    dinv_kernel<<<(N_NODES + 255) / 256, 256, 0, stream>>>(dinv, N_NODES);

    // layer 1
    gemm1_wmma<<<N_NODES / 32, 256, 0, stream>>>(x, W1T, dinv, g1);
    scatter1<<<8192, 256, 0, stream>>>(ei, g1, h1, E);
    finalize1<<<(N_NODES * 128 + 255) / 256, 256, 0, stream>>>(g1, h1, b1, dinv);

    // layer 2
    fill_f32<<<2048, 256, 0, stream>>>(agg2, 0.0f, N_NODES * 40);
    gemm2_wmma<<<N_NODES / 16, 96, 0, stream>>>(h1, W2T, dinv, g2);
    scatter2<<<8192, 256, 0, stream>>>(ei, g2, agg2, E);
    finalize2<<<N_NODES / 8, 256, 0, stream>>>(g2, agg2, b2, dinv, out);
}
